// SNNActDecoder_5299989644007
// MI455X (gfx1250) — matrix operand order repaired and checked by reading, not compiled, not run on hardware
//
#include <hip/hip_runtime.h>

// ---------------------------------------------------------------------------
// SNN actor decoder, algebraically collapsed:
//   out = 2*tanh( s2_p * (S @ w2) + C*(s2_p*(bias2 - m2_p) + b2_p) )
//   S[m,h] = sum_t c_t * spike_t(h_bn1[m,h]),  c_t = (1 - 2^-(16-t))/16
//   h_bn1  = BN1( BN0(x) @ w1 + bias1 )
// GEMM1: v_wmma_f32_16x16x32_bf16, B panel staged into LDS by the Tensor
// Data Mover (tensor_load_to_lds, double-buffered, s_wait_tensorcnt),
// A fragments software-pipelined from global. GEMM2 stays register-direct.
// ---------------------------------------------------------------------------

typedef __attribute__((ext_vector_type(8)))  __bf16       v8bf;
typedef __attribute__((ext_vector_type(16))) __bf16       v16bf;
typedef __attribute__((ext_vector_type(8)))  float        v8f;
typedef __attribute__((ext_vector_type(4)))  unsigned int v4u;
typedef __attribute__((ext_vector_type(8)))  int          v8i;
typedef __attribute__((ext_vector_type(4)))  int          v4i;

#define B_   64
#define P_   16
#define F_   672          // 21 * 32 = 3 * 224
#define H_   2048         // 64 * 32
#define A_   32
#define M_   1024         // B_ * P_
#define EPS_ 1e-5f
#define KC_  224          // K-chunk staged per TDM transfer
#define NCH_ 3            // F_ / KC_

__device__ __forceinline__ unsigned short f2bf(float f) {
  unsigned int u = __float_as_uint(f);
  u += 0x7fffu + ((u >> 16) & 1u);            // round-to-nearest-even
  return (unsigned short)(u >> 16);
}

// A-fragment (16x32 bf16, row-major src, stride ldk elements).
// ISA layout: lanes 0-15 hold row M=lane, K = {0..7, 16..23};
//             lanes 16-31 hold row M=lane-16, K = {8..15, 24..31}.
__device__ __forceinline__ v16bf load_frag_a(const unsigned short* base,
                                             int row, int ldk, int k0, int lane) {
  int kb = k0 + ((lane & 16) ? 8 : 0);
  const unsigned short* p = base + (size_t)row * ldk + kb;
  v8bf lo = *(const v8bf*)(p);
  v8bf hi = *(const v8bf*)(p + 16);
  return __builtin_shufflevector(lo, hi, 0,1,2,3,4,5,6,7,8,9,10,11,12,13,14,15);
}

// B-fragment (32x16 bf16) from transposed weights baseT[N][K] (stride ldk).
// ISA layout: lanes 0-15: col N=lane, K = {0..7, 16..23}; lanes 16-31: col N=lane-16, K = {8..15, 24..31}.
__device__ __forceinline__ v16bf load_frag_b(const unsigned short* baseT,
                                             int col, int ldk, int k0, int lane) {
  int kb = k0 + ((lane & 16) ? 8 : 0);
  const unsigned short* p = baseT + (size_t)col * ldk + kb;
  v8bf lo = *(const v8bf*)(p);
  v8bf hi = *(const v8bf*)(p + 16);
  return __builtin_shufflevector(lo, hi, 0,1,2,3,4,5,6,7,8,9,10,11,12,13,14,15);
}

// Issue a TDM 2D tile copy: 64 rows x KC_ bf16, src row stride F_, packed into
// LDS. D# built per CDNA5 ISA 8.3/8.4 (group0 128b, group1 256b), groups 2/3
// NULL-equivalent zeros for a 2D tile. This toolchain's 6-arg builtin form:
// (uint32x4 g0, int32x8 g1, int32x4, int32x4, int32x8, i32 cpol).
__device__ __forceinline__ void tdm_issue_b(const unsigned short* gsrc,
                                            unsigned lds_off) {
  unsigned long long ga = (unsigned long long)(size_t)gsrc;
  v4u g0;
  g0.x = 1u;                                          // count=1, user mode
  g0.y = lds_off;                                     // lds_addr (bytes)
  g0.z = (unsigned)(ga & 0xffffffffu);                // global_addr[31:0]
  g0.w = (unsigned)((ga >> 32) & 0x01ffffffu)         // global_addr[56:32]
         | (2u << 30);                                // type = 2 ("image")
  v8i g1;
  g1[0] = 0x00010000;                                 // data_size=1 (2 bytes)
  g1[1] = (F_ << 16);                                 // tensor_dim0 lo16 = 672
  g1[2] = (H_ << 16);                                 // dim0 hi=0 | tensor_dim1 lo16
  g1[3] = (KC_ << 16);                                // dim1 hi=0 | tile_dim0 = 224
  g1[4] = 64;                                         // tile_dim1=64 | tile_dim2=0
  g1[5] = F_;                                         // tensor_dim0_stride lo32 = 672
  g1[6] = 0;
  g1[7] = 0;
  v4i z4 = {0, 0, 0, 0};
  v8i z8 = {0, 0, 0, 0, 0, 0, 0, 0};
  __builtin_amdgcn_tensor_load_to_lds(g0, g1, z4, z4, z8, 0);
}

// ---- preprocessing ---------------------------------------------------------

__global__ void snn_bn0_x(const float* __restrict__ x,
                          const float* __restrict__ g, const float* __restrict__ b,
                          const float* __restrict__ m, const float* __restrict__ v,
                          unsigned short* __restrict__ xn) {
  int i = blockIdx.x * blockDim.x + threadIdx.x;     // < M_*F_
  int p = (i / F_) & (P_ - 1);
  float a = rsqrtf(v[p] + EPS_) * g[p];
  xn[i] = f2bf((x[i] - m[p]) * a + b[p]);
}

__global__ void snn_tr_w1(const float* __restrict__ w1, unsigned short* __restrict__ w1t) {
  int i = blockIdx.x * blockDim.x + threadIdx.x;     // < H_*F_
  int n = i / F_, k = i - n * F_;
  w1t[i] = f2bf(w1[(size_t)k * H_ + n]);
}

__global__ void snn_tr_w2(const float* __restrict__ w2, unsigned short* __restrict__ w2t) {
  int i = blockIdx.x * blockDim.x + threadIdx.x;     // < A_*H_
  int a = i / H_, k = i - a * H_;
  w2t[i] = f2bf(w2[(size_t)k * A_ + a]);
}

// ---- GEMM1 (1024x2048x672) + BN1 + 16-step LIF -> weighted spike sum S ----
// Block = 8 waves = 128 rows x 64 cols. B panel (64 x 672) shared by all 8
// waves: TDM-staged into LDS in 3 double-buffered chunks of 64 x 224.

__global__ void __launch_bounds__(256)
snn_gemm1_lif(const unsigned short* __restrict__ xn,
              const unsigned short* __restrict__ w1t,
              const float* __restrict__ bias1,
              const float* __restrict__ g1, const float* __restrict__ b1,
              const float* __restrict__ m1, const float* __restrict__ v1,
              unsigned short* __restrict__ S) {
  __shared__ __align__(16) unsigned short Bpan[2][64 * KC_];   // 2 x 28 KB

  int lane  = threadIdx.x & 31;
  int wave  = threadIdx.x >> 5;
  int mg    = blockIdx.x >> 5;                       // 0..7
  int ng    = blockIdx.x & 31;                       // 0..31
  int mtile = mg * 8 + wave;                         // 0..63
  int n0    = ng * 64;
  int lm    = lane & 15;
  int hi    = (lane >> 4) & 1;

  unsigned lds_base = (unsigned)(size_t)(&Bpan[0][0]);
  const unsigned short* gpanel = w1t + (size_t)n0 * F_;  // 64 rows x 672

  v8f acc[4];
  #pragma unroll
  for (int j = 0; j < 4; ++j)
    acc[j] = (v8f){0.f,0.f,0.f,0.f,0.f,0.f,0.f,0.f};

  int arow = mtile * 16 + lm;

  if (wave == 0) tdm_issue_b(gpanel, lds_base);      // chunk 0 -> buf 0

  for (int c = 0; c < NCH_; ++c) {
    if (wave == 0) {
      if (c + 1 < NCH_) {
        tdm_issue_b(gpanel + (c + 1) * KC_,
                    lds_base + ((unsigned)((c + 1) & 1)) * (64u * KC_ * 2u));
        __builtin_amdgcn_s_wait_tensorcnt(1);        // chunk c landed
      } else {
        __builtin_amdgcn_s_wait_tensorcnt(0);
      }
    }
    __syncthreads();                                  // panel visible to block

    const unsigned short* Bl = &Bpan[c & 1][0];
    int kbase = c * KC_;
    v16bf afrag = load_frag_a(xn, arow, F_, kbase, lane);   // prefetch s=0
    #pragma unroll
    for (int s = 0; s < KC_ / 32; ++s) {
      v16bf acur = afrag;
      if (s < KC_ / 32 - 1)                            // pipeline next A frag
        afrag = load_frag_a(xn, arow, F_, kbase + 32 * (s + 1), lane);
      #pragma unroll
      for (int j = 0; j < 4; ++j) {
        v16bf bf = load_frag_b(Bl, j * 16 + lm, KC_, 32 * s, lane);
        acc[j] = __builtin_amdgcn_wmma_f32_16x16x32_bf16(
            false, acur, false, bf, (short)0, acc[j], false, false);
      }
    }
    __syncthreads();                                  // buf reusable for c+2
  }

  // BN1 constants for the 8 rows this lane owns (p = hi*8 + r).
  float sc[8], of[8];
  #pragma unroll
  for (int r = 0; r < 8; ++r) {
    int p = hi * 8 + r;
    float s = rsqrtf(v1[p] + EPS_) * g1[p];
    sc[r] = s;
    of[r] = b1[p] - m1[p] * s;
  }

  #pragma unroll
  for (int j = 0; j < 4; ++j) {
    int n = n0 + j * 16 + lm;
    float bn = bias1[n];
    #pragma unroll
    for (int r = 0; r < 8; ++r) {
      float hb = (acc[j][r] + bn) * sc[r] + of[r];
      // LIF (tau=2, hard reset, th=1), weighted spike sum in registers.
      float vm = 0.f, Sv = 0.f;
      #pragma unroll
      for (int t = 0; t < 16; ++t) {
        vm = 0.5f * (vm + hb);
        float ct = (1.0f - exp2f((float)t - 16.0f)) * 0.0625f;  // folds at -O3
        bool s = (vm >= 1.0f);
        Sv += s ? ct : 0.0f;
        vm = s ? 0.0f : vm;
      }
      int mrow = mtile * 16 + hi * 8 + r;
      S[(size_t)mrow * H_ + n] = f2bf(Sv);
    }
  }
}

// ---- GEMM2 (1024x32x2048) + BN2/leaky-mean constants + tanh ---------------

__global__ void __launch_bounds__(256)
snn_gemm2_tail(const unsigned short* __restrict__ S,
               const unsigned short* __restrict__ w2t,
               const float* __restrict__ bias2,
               const float* __restrict__ g2, const float* __restrict__ b2,
               const float* __restrict__ m2, const float* __restrict__ v2,
               float* __restrict__ out) {
  int lane  = threadIdx.x & 31;
  int gw    = blockIdx.x * 8 + (threadIdx.x >> 5);   // 0..127
  int mtile = gw >> 1;                               // 0..63
  int n0    = (gw & 1) * 16;
  int lm    = lane & 15;
  int hi    = (lane >> 4) & 1;

  v8f acc = (v8f){0.f,0.f,0.f,0.f,0.f,0.f,0.f,0.f};
  int arow = mtile * 16 + lm;
  int bcol = n0 + lm;
  v16bf af = load_frag_a(S,  arow, H_, 0, lane);
  v16bf bf = load_frag_b(w2t, bcol, H_, 0, lane);
  for (int k0 = 0; k0 < H_; k0 += 32) {
    v16bf ac = af, bc = bf;
    if (k0 + 32 < H_) {                               // pipeline next frags
      af = load_frag_a(S,  arow, H_, k0 + 32, lane);
      bf = load_frag_b(w2t, bcol, H_, k0 + 32, lane);
    }
    acc = __builtin_amdgcn_wmma_f32_16x16x32_bf16(
        false, ac, false, bc, (short)0, acc, false, false);
  }

  const float Cc = 1.0f - (1.0f - 1.0f / 65536.0f) * 0.0625f;  // sum_t c_t
  int n = n0 + lm;
  float bv = bias2[n];
  #pragma unroll
  for (int r = 0; r < 8; ++r) {
    int p = hi * 8 + r;
    float s2  = rsqrtf(v2[p] + EPS_) * g2[p];
    float off = Cc * (s2 * (bv - m2[p]) + b2[p]);
    int mrow = mtile * 16 + hi * 8 + r;
    out[(size_t)mrow * A_ + n] = 2.0f * tanhf(s2 * acc[r] + off);
  }
}

// ---------------------------------------------------------------------------

extern "C" void kernel_launch(void* const* d_in, const int* in_sizes, int n_in,
                              void* d_out, int out_size, void* d_ws, size_t ws_size,
                              hipStream_t stream) {
  (void)in_sizes; (void)n_in; (void)out_size; (void)ws_size;

  const float* x     = (const float*)d_in[0];
  const float* w1    = (const float*)d_in[1];
  const float* bias1 = (const float*)d_in[2];
  const float* w2    = (const float*)d_in[3];
  const float* bias2 = (const float*)d_in[4];
  const float* g0 = (const float*)d_in[5],  *b0 = (const float*)d_in[6];
  const float* m0 = (const float*)d_in[7],  *v0 = (const float*)d_in[8];
  const float* g1 = (const float*)d_in[9],  *b1 = (const float*)d_in[10];
  const float* m1 = (const float*)d_in[11], *v1 = (const float*)d_in[12];
  const float* g2 = (const float*)d_in[13], *b2 = (const float*)d_in[14];
  const float* m2 = (const float*)d_in[15], *v2 = (const float*)d_in[16];

  // Workspace layout (all offsets 256B-aligned by construction).
  char* ws = (char*)d_ws;
  unsigned short* xn  = (unsigned short*)(ws);                       // 1024*672  bf16
  unsigned short* w1t = (unsigned short*)(ws + (size_t)M_ * F_ * 2); // 2048*672  bf16
  unsigned short* w2t = (unsigned short*)(ws + (size_t)(M_ + H_) * F_ * 2);          // 32*2048
  unsigned short* Sm  = (unsigned short*)(ws + (size_t)(M_ + H_) * F_ * 2
                                             + (size_t)A_ * H_ * 2);                  // 1024*2048

  snn_bn0_x <<<(M_ * F_) / 256, 256, 0, stream>>>(x, g0, b0, m0, v0, xn);
  snn_tr_w1 <<<(H_ * F_) / 256, 256, 0, stream>>>(w1, w1t);
  snn_tr_w2 <<<(A_ * H_) / 256, 256, 0, stream>>>(w2, w2t);

  // 256 blocks * 8 waves: each block 128 rows x 64 cols, TDM-staged B panel.
  snn_gemm1_lif <<<256, 256, 0, stream>>>(xn, w1t, bias1, g1, b1, m1, v1, Sm);

  // 128 wave-tiles (16x16 each) = 16 blocks * 8 waves
  snn_gemm2_tail <<<16, 256, 0, stream>>>(Sm, w2t, bias2, g2, b2, m2, v2,
                                          (float*)d_out);
}
